// sd3_adapter_attn_processor_79860621902020
// MI455X (gfx1250) — matrix operand order repaired and checked by reading, not compiled, not run on hardware
//
#include <hip/hip_runtime.h>
#include <hip/hip_bf16.h>
#include <math.h>

// ---------------------------------------------------------------------------
// Problem constants (from reference)
// ---------------------------------------------------------------------------
#define NQv    2048
#define NTv    154
#define NPv    256
#define Dv     1536
#define Hv     24
#define HDv    64
#define PICSv  1024
#define NTOTv  2202          // NQ + NT
#define NTOTP  2304          // padded to multiple of 128
#define NTP    160           // encoder rows padded to multiple of 32
#define SCALEv 0.125f
#define BLENDv (0.75f * 0.65f)

typedef __bf16 bf16_t;
typedef __attribute__((ext_vector_type(16))) __bf16 v16bf;
typedef __attribute__((ext_vector_type(8)))  __bf16 v8bf;
typedef __attribute__((ext_vector_type(8)))  float  v8f;

__device__ __forceinline__ v8f zero8() {
    v8f z;
#pragma unroll
    for (int i = 0; i < 8; ++i) z[i] = 0.f;
    return z;
}

// A-fragment (16x32 bf16, M x K): lane<16 holds row=lane, K {0..7,16..23};
// lane>=16 holds row=lane-16, K {8..15,24..31}. Caller pre-offsets by (hi*8).
__device__ __forceinline__ v16bf load_a_frag(const bf16_t* p) {
    union { v16bf v; v8bf h[2]; } u;
    u.h[0] = *(const v8bf*)(p);
    u.h[1] = *(const v8bf*)(p + 16);
    return u.v;
}
// B-fragment (32x16 bf16, K x N): lane<16 holds col=lane, K {0..7,16..23};
// lane>=16 holds col=lane-16, K {8..15,24..31}. Caller pre-offsets by (hi*8).
__device__ __forceinline__ v16bf load_b_frag(const bf16_t* p) {
    union { v16bf v; v8bf h[2]; } u;
    u.h[0] = *(const v8bf*)(p);
    u.h[1] = *(const v8bf*)(p + 16);
    return u.v;
}

__device__ __forceinline__ v8f wmma_bf16(v16bf a, v16bf b, v8f c) {
    return __builtin_amdgcn_wmma_f32_16x16x32_bf16(
        /*neg_a=*/false, a, /*neg_b=*/false, b,
        /*c_mod=*/(short)0, c, /*reuse_a=*/false, /*reuse_b=*/false);
}

// ---------------------------------------------------------------------------
// Elementwise helpers
// ---------------------------------------------------------------------------
__global__ void cvt_pad_kernel(const float* __restrict__ src, bf16_t* __restrict__ dst,
                               int Mreal, int K, long total) {
    long i = (long)blockIdx.x * blockDim.x + threadIdx.x;
    if (i >= total) return;
    long row = i / K;
    dst[i] = (row < Mreal) ? (bf16_t)src[i] : (bf16_t)0.f;
}

// Wt[n*K + k] = W[k*N + n]  (weights stored transposed for B-fragment loads)
__global__ void transpose_w_kernel(const float* __restrict__ W, bf16_t* __restrict__ Wt,
                                   int K, int N, long total) {
    long i = (long)blockIdx.x * blockDim.x + threadIdx.x;
    if (i >= total) return;
    long n = i / K, k = i % K;
    Wt[i] = (bf16_t)W[k * (long)N + n];
}

// Xh[h][r][d] = X[r][h*64+d]  (row-major per-head, padded rows zeroed)
__global__ void pack_heads_kernel(const float* __restrict__ X, bf16_t* __restrict__ Xh,
                                  int Rreal, int Rp) {
    long total = (long)Hv * Rp * HDv;
    long i = (long)blockIdx.x * blockDim.x + threadIdx.x;
    if (i >= total) return;
    int d = (int)(i % HDv);
    long t = i / HDv;
    int r = (int)(t % Rp);
    int h = (int)(t / Rp);
    Xh[i] = (r < Rreal) ? (bf16_t)X[(long)r * Dv + h * HDv + d] : (bf16_t)0.f;
}

// XT[h][d][r] = X[r][h*64+d]  (transposed per-head for B-fragments of P@V)
__global__ void pack_headsT_kernel(const float* __restrict__ X, bf16_t* __restrict__ XT,
                                   int Rreal, int Rp) {
    long total = (long)Hv * HDv * Rp;
    long i = (long)blockIdx.x * blockDim.x + threadIdx.x;
    if (i >= total) return;
    int r = (int)(i % Rp);
    long t = i / Rp;
    int d = (int)(t % HDv);
    int h = (int)(t / HDv);
    XT[i] = (r < Rreal) ? (bf16_t)X[(long)r * Dv + h * HDv + d] : (bf16_t)0.f;
}

// dst[r][c] = src[row_off + r][c] for r < Mreal, else 0 (padded rows)
__global__ void pack_sub_kernel(const float* __restrict__ src, bf16_t* __restrict__ dst,
                                int row_off, int Mreal, int Mp, int K) {
    long total = (long)Mp * K;
    long i = (long)blockIdx.x * blockDim.x + threadIdx.x;
    if (i >= total) return;
    long row = i / K, col = i % K;
    dst[i] = (row < Mreal) ? (bf16_t)src[(row_off + row) * (long)K + col] : (bf16_t)0.f;
}

// ---------------------------------------------------------------------------
// Generic bf16 WMMA GEMM:  C(MxN,f32) = A(MxK,bf16) * Bt(NxK,bf16)^T + bias
// One wave -> 32x64 tile (2 M-tiles x 4 N-tiles). Block = 4 waves (stacked M).
// ---------------------------------------------------------------------------
__global__ __launch_bounds__(128) void gemm_bf16_kernel(
    const bf16_t* __restrict__ A, const bf16_t* __restrict__ Bt,
    const float* __restrict__ bias, float* __restrict__ C,
    int Mreal, int Mp, int N, int K) {
    const int lane = threadIdx.x & 31;
    const int wave = threadIdx.x >> 5;
    const int m0 = (blockIdx.y * 4 + wave) * 32;
    const int n0 = blockIdx.x * 64;
    if (m0 >= Mp) return;               // uniform per wave -> EXEC all-ones for WMMA
    const int hi = lane >> 4;
    const int ln = lane & 15;

    v8f acc[2][4];
#pragma unroll
    for (int m = 0; m < 2; ++m)
#pragma unroll
        for (int t = 0; t < 4; ++t) acc[m][t] = zero8();

    const bf16_t* Ar0 = A + (size_t)(m0 + ln) * K + hi * 8;
    const bf16_t* Ar1 = A + (size_t)(m0 + 16 + ln) * K + hi * 8;

    for (int k0 = 0; k0 < K; k0 += 32) {
        v16bf a0 = load_a_frag(Ar0 + k0);
        v16bf a1 = load_a_frag(Ar1 + k0);
#pragma unroll
        for (int t = 0; t < 4; ++t) {
            v16bf b = load_b_frag(Bt + (size_t)(n0 + t * 16 + ln) * K + k0 + hi * 8);
            acc[0][t] = wmma_bf16(a0, b, acc[0][t]);
            acc[1][t] = wmma_bf16(a1, b, acc[1][t]);
        }
    }

#pragma unroll
    for (int t = 0; t < 4; ++t) {
        const int col = n0 + t * 16 + ln;
        const float bv = bias ? bias[col] : 0.f;
#pragma unroll
        for (int m = 0; m < 2; ++m)
#pragma unroll
            for (int j = 0; j < 8; ++j) {
                int row = m0 + m * 16 + j + hi * 8;
                if (row < Mreal) C[(size_t)row * N + col] = acc[m][t][j] + bv;
            }
    }
}

// ---------------------------------------------------------------------------
// Attention 1 (pics cross-attention with clip gate). Block = 2 waves, each
// wave handles 16 query rows; full 256-key row held in LDS (gate needs it).
// ---------------------------------------------------------------------------
__global__ __launch_bounds__(64) void attn1_kernel(
    const bf16_t* __restrict__ Qh,   // [H][NTOTP][64] (rows < 2048 used)
    const bf16_t* __restrict__ Kh,   // [H][256][64]
    const bf16_t* __restrict__ VhT,  // [H][64][256]
    const float* __restrict__ clip,  // [256]
    const float* __restrict__ Wsl,   // [512]
    const float* __restrict__ bsl,   // [1]
    float* __restrict__ multi)       // [2048][1536]
{
    __shared__ __align__(32) float  S[2][16 * 256];
    __shared__ __align__(32) bf16_t P[2][16 * 256];
    __shared__ float Lrow[2][16];

    const int lane = threadIdx.x & 31;
    const int wave = threadIdx.x >> 5;
    const int hi = lane >> 4, ln = lane & 15;
    const int h = blockIdx.x;
    const int q0 = blockIdx.y * 32 + wave * 16;
    float*  Sw = S[wave];
    bf16_t* Pw = P[wave];
    float*  Lw = Lrow[wave];

    // scalar gate term: clip @ Wsl[NP:] + bsl
    float cs = 0.f;
    for (int k = lane; k < NPv; k += 32) cs += clip[k] * Wsl[NPv + k];
#pragma unroll
    for (int off = 16; off >= 1; off >>= 1) cs += __shfl_xor(cs, off);
    cs += bsl[0];

    const bf16_t* qp = Qh + ((size_t)h * NTOTP + q0 + ln) * HDv + hi * 8;
    v16bf a0 = load_a_frag(qp);
    v16bf a1 = load_a_frag(qp + 32);

    // scores: Q(16x64) @ K^T(64x256) -> LDS (scaled)
#pragma unroll
    for (int nt = 0; nt < 16; ++nt) {
        const bf16_t* kp = Kh + ((size_t)h * NPv + nt * 16 + ln) * HDv + hi * 8;
        v8f acc = zero8();
        acc = wmma_bf16(a0, load_b_frag(kp), acc);
        acc = wmma_bf16(a1, load_b_frag(kp + 32), acc);
#pragma unroll
        for (int j = 0; j < 8; ++j)
            Sw[(j + hi * 8) * 256 + nt * 16 + ln] = acc[j] * SCALEv;
    }
    __syncthreads();

    // gate + softmax: 2 lanes per row, 128 cols each
    {
        const int row = lane >> 1, cb = (lane & 1) * 128;
        float gp = 0.f;
        for (int i = 0; i < 128; ++i) gp += Sw[row * 256 + cb + i] * Wsl[cb + i];
        gp += __shfl_xor(gp, 1);
        const float gate = gp + cs;
        float mx = -1e30f;
        for (int i = 0; i < 128; ++i) {
            float l = Sw[row * 256 + cb + i] + gate * clip[cb + i];
            mx = fmaxf(mx, l);
        }
        mx = fmaxf(mx, __shfl_xor(mx, 1));
        float sum = 0.f;
        for (int i = 0; i < 128; ++i) {
            float l = Sw[row * 256 + cb + i] + gate * clip[cb + i];
            float p = __expf(l - mx);
            sum += p;
            Pw[row * 256 + cb + i] = (bf16_t)p;
        }
        sum += __shfl_xor(sum, 1);
        if ((lane & 1) == 0) Lw[row] = sum;
    }
    __syncthreads();

    // multi tile = P(16x256) @ V(256x64)
    v8f O[4];
#pragma unroll
    for (int t = 0; t < 4; ++t) O[t] = zero8();
#pragma unroll
    for (int ks = 0; ks < 8; ++ks) {
        v16bf ap = load_a_frag(&Pw[ln * 256 + ks * 32 + hi * 8]);
#pragma unroll
        for (int t = 0; t < 4; ++t) {
            const bf16_t* vp = VhT + ((size_t)h * HDv + t * 16 + ln) * NPv + ks * 32 + hi * 8;
            O[t] = wmma_bf16(ap, load_b_frag(vp), O[t]);
        }
    }
    float lv[8];
#pragma unroll
    for (int j = 0; j < 8; ++j) lv[j] = Lw[j + hi * 8];
#pragma unroll
    for (int t = 0; t < 4; ++t)
#pragma unroll
        for (int j = 0; j < 8; ++j)
            multi[(size_t)(q0 + j + hi * 8) * Dv + h * HDv + t * 16 + ln] = O[t][j] / lv[j];
}

// ---------------------------------------------------------------------------
// Attention 2 (self-attn over 2202 tokens), flash-style online softmax.
// Block = 4 waves x 16 query rows; 128-key chunks; O in 32 accumulator VGPRs.
// ---------------------------------------------------------------------------
__global__ __launch_bounds__(128) void attn2_kernel(
    const bf16_t* __restrict__ Qh,  // [H][2304][64]
    const bf16_t* __restrict__ Kh,  // [H][2304][64]
    const bf16_t* __restrict__ VT,  // [H][64][2304]
    float* __restrict__ hs)         // [2304][1536]
{
    __shared__ __align__(32) float  S[4][16 * 128];
    __shared__ __align__(32) bf16_t P[4][16 * 128];
    __shared__ float Mrow[4][16], Lrow[4][16], Rrow[4][16];

    const int lane = threadIdx.x & 31;
    const int wave = threadIdx.x >> 5;
    const int hi = lane >> 4, ln = lane & 15;
    const int h = blockIdx.x;
    const int q0 = blockIdx.y * 64 + wave * 16;
    float*  Sw = S[wave];
    bf16_t* Pw = P[wave];
    float*  Mw = Mrow[wave];
    float*  Lw = Lrow[wave];
    float*  Rw = Rrow[wave];

    if (lane < 16) { Mw[lane] = -1e30f; Lw[lane] = 0.f; }

    const bf16_t* qp = Qh + ((size_t)h * NTOTP + q0 + ln) * HDv + hi * 8;
    v16bf a0 = load_a_frag(qp);
    v16bf a1 = load_a_frag(qp + 32);
    v8f O[4];
#pragma unroll
    for (int t = 0; t < 4; ++t) O[t] = zero8();
    __syncthreads();

    for (int c = 0; c < NTOTP / 128; ++c) {
        const int k0 = c * 128;
        // score chunk: Q(16x64) @ K^T(64x128) -> LDS, scale + mask
#pragma unroll
        for (int nt = 0; nt < 8; ++nt) {
            const bf16_t* kp = Kh + ((size_t)h * NTOTP + k0 + nt * 16 + ln) * HDv + hi * 8;
            v8f acc = zero8();
            acc = wmma_bf16(a0, load_b_frag(kp), acc);
            acc = wmma_bf16(a1, load_b_frag(kp + 32), acc);
            const int col = nt * 16 + ln;
            const float mask = (k0 + col >= NTOTv) ? -1e30f : 0.f;
#pragma unroll
            for (int j = 0; j < 8; ++j) {
                float v = acc[j] * SCALEv + mask;
                Sw[(j + hi * 8) * 128 + col] = v;
            }
        }
        __syncthreads();

        // online softmax stats: 2 lanes per row, 64 cols each
        {
            const int row = lane >> 1, cb = (lane & 1) * 64;
            float mx = -1e30f;
            for (int i = 0; i < 64; ++i) mx = fmaxf(mx, Sw[row * 128 + cb + i]);
            mx = fmaxf(mx, __shfl_xor(mx, 1));
            const float mold = Mw[row];
            const float mnew = fmaxf(mold, mx);
            float sum = 0.f;
            for (int i = 0; i < 64; ++i) {
                float p = __expf(Sw[row * 128 + cb + i] - mnew);
                sum += p;
                Pw[row * 128 + cb + i] = (bf16_t)p;
            }
            sum += __shfl_xor(sum, 1);
            if ((lane & 1) == 0) {
                Rw[row] = __expf(mold - mnew);
                Mw[row] = mnew;
                Lw[row] = Lw[row] * Rw[row] + sum;
            }
        }
        __syncthreads();

        // rescale O, then O += P(16x128) @ V(128x64)
        float rv[8];
#pragma unroll
        for (int j = 0; j < 8; ++j) rv[j] = Rw[j + hi * 8];
#pragma unroll
        for (int t = 0; t < 4; ++t)
#pragma unroll
            for (int j = 0; j < 8; ++j) O[t][j] *= rv[j];
#pragma unroll
        for (int ks = 0; ks < 4; ++ks) {
            v16bf ap = load_a_frag(&Pw[ln * 128 + ks * 32 + hi * 8]);
#pragma unroll
            for (int t = 0; t < 4; ++t) {
                const bf16_t* vp = VT + ((size_t)h * HDv + t * 16 + ln) * NTOTP
                                      + k0 + ks * 32 + hi * 8;
                O[t] = wmma_bf16(ap, load_b_frag(vp), O[t]);
            }
        }
        __syncthreads();
    }

    float lv[8];
#pragma unroll
    for (int j = 0; j < 8; ++j) lv[j] = Lw[j + hi * 8];
#pragma unroll
    for (int t = 0; t < 4; ++t)
#pragma unroll
        for (int j = 0; j < 8; ++j)
            hs[(size_t)(q0 + j + hi * 8) * Dv + h * HDv + t * 16 + ln] = O[t][j] / lv[j];
}

// ---------------------------------------------------------------------------
// adaptive = sigmoid(multi @ Wad[:D] + hs_img @ Wad[D:] + bad), one block per q
// ---------------------------------------------------------------------------
__global__ __launch_bounds__(256) void adaptive_kernel(
    const float* __restrict__ multi, const float* __restrict__ hs,
    const float* __restrict__ Wad, const float* __restrict__ bad,
    float* __restrict__ adaptv) {
    __shared__ float red[256];
    const int q = blockIdx.x, t = threadIdx.x;
    float p = 0.f;
    for (int i = t; i < Dv; i += 256)
        p += multi[(size_t)q * Dv + i] * Wad[i] + hs[(size_t)q * Dv + i] * Wad[Dv + i];
    red[t] = p;
    __syncthreads();
    for (int s = 128; s > 0; s >>= 1) {
        if (t < s) red[t] += red[t + s];
        __syncthreads();
    }
    if (t == 0) adaptv[q] = 1.f / (1.f + __expf(-(red[0] + bad[0])));
}

__global__ void blend_kernel(const float* __restrict__ multi, const float* __restrict__ hs,
                             const float* __restrict__ adaptv, bf16_t* __restrict__ out) {
    long total = (long)NQv * Dv;
    long i = (long)blockIdx.x * blockDim.x + threadIdx.x;
    if (i >= total) return;
    long q = i / Dv;
    out[i] = (bf16_t)(BLENDv * adaptv[q] * multi[i] + hs[i]);
}

// ---------------------------------------------------------------------------
// Host orchestration
// ---------------------------------------------------------------------------
extern "C" void kernel_launch(void* const* d_in, const int* in_sizes, int n_in,
                              void* d_out, int out_size, void* d_ws, size_t ws_size,
                              hipStream_t stream) {
    (void)in_sizes; (void)n_in; (void)out_size; (void)ws_size;
    const float* hidden = (const float*)d_in[0];
    const float* enc    = (const float*)d_in[1];
    const float* pics   = (const float*)d_in[2];
    const float* clip   = (const float*)d_in[3];
    const float* Wq  = (const float*)d_in[4];  const float* bq   = (const float*)d_in[5];
    const float* Wk  = (const float*)d_in[6];  const float* bk   = (const float*)d_in[7];
    const float* Wv  = (const float*)d_in[8];  const float* bv   = (const float*)d_in[9];
    const float* Waq = (const float*)d_in[10]; const float* baq  = (const float*)d_in[11];
    const float* Wak = (const float*)d_in[12]; const float* bak  = (const float*)d_in[13];
    const float* Wav = (const float*)d_in[14]; const float* bav  = (const float*)d_in[15];
    const float* Wkl = (const float*)d_in[16]; const float* Wvl  = (const float*)d_in[17];
    const float* Wsl = (const float*)d_in[18]; const float* bsl  = (const float*)d_in[19];
    const float* Wad = (const float*)d_in[20]; const float* bad  = (const float*)d_in[21];
    const float* Wout= (const float*)d_in[22]; const float* bout = (const float*)d_in[23];
    const float* Wadd= (const float*)d_in[24]; const float* badd = (const float*)d_in[25];
    float* out = (float*)d_out;

    char* ws = (char*)d_ws;
    size_t off = 0;
    auto alloc = [&](size_t bytes) -> void* {
        void* p = ws + off;
        off += (bytes + 255) & ~(size_t)255;
        return p;
    };

    bf16_t* hb   = (bf16_t*)alloc((size_t)NQv * Dv * 2);
    bf16_t* eb   = (bf16_t*)alloc((size_t)NTP * Dv * 2);
    bf16_t* pb   = (bf16_t*)alloc((size_t)NPv * PICSv * 2);
    bf16_t* WqT  = (bf16_t*)alloc((size_t)Dv * Dv * 2);
    bf16_t* WkT  = (bf16_t*)alloc((size_t)Dv * Dv * 2);
    bf16_t* WvT  = (bf16_t*)alloc((size_t)Dv * Dv * 2);
    bf16_t* WaqT = (bf16_t*)alloc((size_t)Dv * Dv * 2);
    bf16_t* WakT = (bf16_t*)alloc((size_t)Dv * Dv * 2);
    bf16_t* WavT = (bf16_t*)alloc((size_t)Dv * Dv * 2);
    bf16_t* WoutT= (bf16_t*)alloc((size_t)Dv * Dv * 2);
    bf16_t* WaddT= (bf16_t*)alloc((size_t)Dv * Dv * 2);
    bf16_t* WklT = (bf16_t*)alloc((size_t)Dv * PICSv * 2);
    bf16_t* WvlT = (bf16_t*)alloc((size_t)Dv * PICSv * 2);
    float*  q2   = (float*)alloc((size_t)NTOTP * Dv * 4);
    float*  k2   = (float*)alloc((size_t)NTOTP * Dv * 4);
    float*  v2   = (float*)alloc((size_t)NTOTP * Dv * 4);
    bf16_t* Q2h  = (bf16_t*)alloc((size_t)Hv * NTOTP * HDv * 2);
    bf16_t* K2h  = (bf16_t*)alloc((size_t)Hv * NTOTP * HDv * 2);
    bf16_t* V2T  = (bf16_t*)alloc((size_t)Hv * HDv * NTOTP * 2);
    float*  Kp   = (float*)alloc((size_t)NPv * Dv * 4);
    float*  Vp   = (float*)alloc((size_t)NPv * Dv * 4);
    bf16_t* Kh1  = (bf16_t*)alloc((size_t)Hv * NPv * HDv * 2);
    bf16_t* VT1  = (bf16_t*)alloc((size_t)Hv * HDv * NPv * 2);
    float*  multi= (float*)alloc((size_t)NQv * Dv * 4);
    float*  hs   = (float*)alloc((size_t)NTOTP * Dv * 4);
    float*  adapt= (float*)alloc((size_t)NQv * 4);
    bf16_t* blnd = (bf16_t*)alloc((size_t)NQv * Dv * 2);
    bf16_t* txtb = (bf16_t*)alloc((size_t)NTP * Dv * 2);

    auto cvt = [&](const float* s, bf16_t* d, int Mreal, int K, int Mp) {
        long total = (long)Mp * K;
        cvt_pad_kernel<<<dim3((unsigned)((total + 255) / 256)), dim3(256), 0, stream>>>(
            s, d, Mreal, K, total);
    };
    auto trw = [&](const float* W, bf16_t* Wt, int K, int N) {
        long total = (long)N * K;
        transpose_w_kernel<<<dim3((unsigned)((total + 255) / 256)), dim3(256), 0, stream>>>(
            W, Wt, K, N, total);
    };
    auto gemm = [&](const bf16_t* A, const bf16_t* Bt, const float* bias, float* C,
                    int Mreal, int Mp, int N, int K) {
        dim3 grid(N / 64, (Mp + 127) / 128);
        gemm_bf16_kernel<<<grid, dim3(128), 0, stream>>>(A, Bt, bias, C, Mreal, Mp, N, K);
    };

    // 1) activations -> bf16 (padded)
    cvt(hidden, hb, NQv, Dv, NQv);
    cvt(enc,    eb, NTv, Dv, NTP);
    cvt(pics,   pb, NPv, PICSv, NPv);

    // 2) weights -> bf16 transposed
    trw(Wq, WqT, Dv, Dv);    trw(Wk, WkT, Dv, Dv);    trw(Wv, WvT, Dv, Dv);
    trw(Waq, WaqT, Dv, Dv);  trw(Wak, WakT, Dv, Dv);  trw(Wav, WavT, Dv, Dv);
    trw(Wout, WoutT, Dv, Dv);trw(Wadd, WaddT, Dv, Dv);
    trw(Wkl, WklT, PICSv, Dv); trw(Wvl, WvlT, PICSv, Dv);

    // 3) projections (q2/k2/v2 = concat(image proj, encoder proj))
    gemm(hb, WqT,  bq,  q2,                      NQv, NQv, Dv, Dv);
    gemm(eb, WaqT, baq, q2 + (size_t)NQv * Dv,   NTv, NTP, Dv, Dv);
    gemm(hb, WkT,  bk,  k2,                      NQv, NQv, Dv, Dv);
    gemm(eb, WakT, bak, k2 + (size_t)NQv * Dv,   NTv, NTP, Dv, Dv);
    gemm(hb, WvT,  bv,  v2,                      NQv, NQv, Dv, Dv);
    gemm(eb, WavT, bav, v2 + (size_t)NQv * Dv,   NTv, NTP, Dv, Dv);
    gemm(pb, WklT, nullptr, Kp, NPv, NPv, Dv, PICSv);
    gemm(pb, WvlT, nullptr, Vp, NPv, NPv, Dv, PICSv);

    // 4) split heads (padded, bf16)
    {
        long t1 = (long)Hv * NTOTP * HDv;
        pack_heads_kernel<<<dim3((unsigned)((t1 + 255) / 256)), dim3(256), 0, stream>>>(
            q2, Q2h, NTOTv, NTOTP);
        pack_heads_kernel<<<dim3((unsigned)((t1 + 255) / 256)), dim3(256), 0, stream>>>(
            k2, K2h, NTOTv, NTOTP);
        pack_headsT_kernel<<<dim3((unsigned)((t1 + 255) / 256)), dim3(256), 0, stream>>>(
            v2, V2T, NTOTv, NTOTP);
        long t2 = (long)Hv * NPv * HDv;
        pack_heads_kernel<<<dim3((unsigned)((t2 + 255) / 256)), dim3(256), 0, stream>>>(
            Kp, Kh1, NPv, NPv);
        pack_headsT_kernel<<<dim3((unsigned)((t2 + 255) / 256)), dim3(256), 0, stream>>>(
            Vp, VT1, NPv, NPv);
    }

    // 5) attention 1 (pics cross-attn with gate) -> multi
    attn1_kernel<<<dim3(Hv, NQv / 32), dim3(64), 0, stream>>>(
        Q2h, Kh1, VT1, clip, Wsl, bsl, multi);

    // 6) attention 2 (flash self-attn over 2202 tokens) -> hs
    attn2_kernel<<<dim3(Hv, (NTOTv + 63) / 64), dim3(128), 0, stream>>>(
        Q2h, K2h, V2T, hs);

    // 7) adaptive gate + blend
    adaptive_kernel<<<dim3(NQv), dim3(256), 0, stream>>>(multi, hs, Wad, bad, adapt);
    {
        long total = (long)NQv * Dv;
        blend_kernel<<<dim3((unsigned)((total + 255) / 256)), dim3(256), 0, stream>>>(
            multi, hs, adapt, blnd);
    }

    // 8) output projections straight into d_out
    gemm(blnd, WoutT, bout, out, NQv, NQv, Dv, Dv);
    {
        long total = (long)NTP * Dv;
        pack_sub_kernel<<<dim3((unsigned)((total + 255) / 256)), dim3(256), 0, stream>>>(
            hs, txtb, NQv, NTv, NTP, Dv);
    }
    gemm(txtb, WaddT, badd, out + (size_t)NQv * Dv, NTv, NTP, Dv, Dv);
}
